// GridGNN_17643725652557
// MI455X (gfx1250) — matrix operand order repaired and checked by reading, not compiled, not run on hardware
//
#include <hip/hip_runtime.h>
#include <hip/hip_bf16.h>

#define HCd 128
#define Hh 4
#define Cc 32
#define FE 16
#define NGRP 64
#define NCLS 10

typedef __attribute__((ext_vector_type(16))) __bf16 v16bf;
typedef __attribute__((ext_vector_type(16))) unsigned short v16us;
typedef __attribute__((ext_vector_type(8))) float v8f;
typedef __attribute__((ext_vector_type(8))) unsigned int v8u;

static __device__ __forceinline__ unsigned short f2bf(float f) {
  unsigned u = __builtin_bit_cast(unsigned, f);
  u += 0x7FFFu + ((u >> 16) & 1u);          // round-to-nearest-even
  return (unsigned short)(u >> 16);
}
static __device__ __forceinline__ unsigned encf(float f) {  // order-preserving f32->u32
  unsigned u = __builtin_bit_cast(unsigned, f);
  return ((int)u >= 0) ? (u | 0x80000000u) : ~u;
}
static __device__ __forceinline__ float decf(unsigned u) {
  unsigned v = (u & 0x80000000u) ? (u & 0x7FFFFFFFu) : ~u;
  return __builtin_bit_cast(float, v);
}
static __device__ __forceinline__ void atomAddF(float* p, float v) {
  __hip_atomic_fetch_add(p, v, __ATOMIC_RELAXED, __HIP_MEMORY_SCOPE_AGENT);
}
static __device__ __forceinline__ float lrelu(float x) { return x >= 0.f ? x : 0.2f * x; }

// ---------------- utility ----------------
__global__ void k_zero(float* p, int n) {
  int i = blockIdx.x * blockDim.x + threadIdx.x;
  if (i < n) p[i] = 0.f;
}

// convert f32 weight [KIN x KOUT] -> bf16 stored TRANSPOSED [KOUT x KIN]
__global__ void k_wcvt_t(const float* __restrict__ w, unsigned short* __restrict__ o,
                         int KIN, int KOUT) {
  int i = blockIdx.x * blockDim.x + threadIdx.x;
  if (i >= KIN * KOUT) return;
  int k = i / KOUT, c = i - k * KOUT;
  o[c * KIN + k] = f2bf(w[i]);
}

// column sums of edge_attr [E x 16]
__global__ void k_colsum16(const float* __restrict__ ea, float* __restrict__ sums, int E) {
  __shared__ float s[FE];
  if (threadIdx.x < FE) s[threadIdx.x] = 0.f;
  __syncthreads();
  int stride = gridDim.x * blockDim.x;
  for (int i = blockIdx.x * blockDim.x + threadIdx.x; i < E * FE; i += stride)
    atomicAdd(&s[i & 15], ea[i]);
  __syncthreads();
  if (threadIdx.x < FE) atomAddF(&sums[threadIdx.x], s[threadIdx.x]);
}

// per-layer fold: M[f][h] = sum_c We[f, h*C+c]*aE[h,c];  le_self[h] = mean(ea) . M[:,h]
__global__ void k_prep(const float* __restrict__ Wep, const float* __restrict__ aEp,
                       const float* __restrict__ meanS, float* __restrict__ M,
                       float* __restrict__ leself, int E) {
  int t = threadIdx.x;               // 64 threads
  int f = t >> 2, h = t & 3;
  float m = 0.f;
  for (int c = 0; c < Cc; ++c) m += Wep[f * HCd + h * Cc + c] * aEp[h * Cc + c];
  M[f * Hh + h] = m;
  __syncthreads();
  if (t < Hh) {
    float s = 0.f, invE = 1.0f / (float)E;
    for (int ff = 0; ff < FE; ++ff) s += meanS[ff] * invE * M[ff * Hh + t];
    leself[t] = s;
  }
}

// ---------------- WMMA GEMM:  C[nrows x KOUT] = act(A[nrows x KIN] @ W + bias) ----------------
// WbfT is the bf16 weight stored transposed: element (k, col) at WbfT[col*KIN + k].
template <int KIN, int KOUT, bool BIAS, bool RELU>
__launch_bounds__(256)
__global__ void k_gemm(const float* __restrict__ A, const unsigned short* __restrict__ WbfT,
                       const float* __restrict__ bias, float* __restrict__ Cout, int nrows) {
  const int wave = threadIdx.x >> 5;
  const int lane = threadIdx.x & 31;
  const int hf = lane >> 4;          // half-wave
  const int l15 = lane & 15;
  const int rowBase = blockIdx.x * 16;
  const int colBase = wave * 16;
  int m = rowBase + l15;             // row this lane supplies for the A fragment
  if (m > nrows - 1) m = nrows - 1;  // clamp: garbage rows only feed masked D rows
  const int col = colBase + l15;
  const float* __restrict__ arow = A + (size_t)m * KIN;
  const unsigned short* __restrict__ wcol = WbfT + (size_t)col * KIN;

  v8f acc = {};
#pragma unroll
  for (int k0 = 0; k0 < KIN; k0 += 32) {
    // A 16x32 bf16: lane holds 16 contiguous K (k = 16*hf + s), matching B
    v8f alo = *(const v8f*)(arow + k0 + 16 * hf);
    v8f ahi = *(const v8f*)(arow + k0 + 16 * hf + 8);
    v16us au;
#pragma unroll
    for (int j = 0; j < 8; ++j) { au[j] = f2bf(alo[j]); au[j + 8] = f2bf(ahi[j]); }
    // B 32x16 bf16: lane holds 16 contiguous K of its column (transposed storage)
    v8u braw = *(const v8u*)(wcol + k0 + 16 * hf);
    v16bf a = __builtin_bit_cast(v16bf, au);
    v16bf b = __builtin_bit_cast(v16bf, braw);
    acc = __builtin_amdgcn_wmma_f32_16x16x32_bf16(false, a, false, b, (short)0, acc, false, false);
  }

  // C/D layout: row = rowBase + 8*hf + r, col = colBase + (lane&15)
  if (rowBase + 16 <= nrows) {
#pragma unroll
    for (int r = 0; r < 8; ++r) {
      float v = acc[r];
      if (BIAS) v += bias[col];
      if (RELU) v = fmaxf(v, 0.f);
      Cout[(size_t)(rowBase + 8 * hf + r) * KOUT + col] = v;
    }
  } else {
#pragma unroll
    for (int r = 0; r < 8; ++r) {
      int row = rowBase + 8 * hf + r;
      if (row < nrows) {
        float v = acc[r];
        if (BIAS) v += bias[col];
        if (RELU) v = fmaxf(v, 0.f);
        Cout[(size_t)row * KOUT + col] = v;
      }
    }
  }
}

// ---------------- per-node alpha + self-loop max ----------------
__global__ void k_node_alpha(const float* __restrict__ hp, const float* __restrict__ aSp,
                             const float* __restrict__ aDp, const float* __restrict__ leself,
                             float* __restrict__ lsb, float* __restrict__ ldb,
                             unsigned* __restrict__ mEnc, int N) {
  int t = blockIdx.x * blockDim.x + threadIdx.x;
  if (t >= N * Hh) return;
  int n = t >> 2, h = t & 3;
  const float* r = hp + (size_t)n * HCd + h * Cc;
  const float* as = aSp + h * Cc;
  const float* ad = aDp + h * Cc;
  float s = 0.f, d = 0.f;
#pragma unroll 8
  for (int c = 0; c < Cc; ++c) { s += r[c] * as[c]; d += r[c] * ad[c]; }
  lsb[t] = s; ldb[t] = d;
  atomicMax(&mEnc[t], encf(lrelu(s + d + leself[h])));
}

// ---------------- edge pass A: le + logit max ----------------
__global__ void k_edge_max(const int* __restrict__ src, const int* __restrict__ dst,
                           const float* __restrict__ ea, const float* __restrict__ M,
                           const float* __restrict__ lsb, const float* __restrict__ ldb,
                           float* __restrict__ lebuf, unsigned* __restrict__ mEnc, int E) {
  int e = blockIdx.x * blockDim.x + threadIdx.x;
  if (e >= E) return;
  float eav[FE];
  const float* er = ea + (size_t)e * FE;
#pragma unroll
  for (int f = 0; f < FE; ++f) eav[f] = er[f];
  int s = src[e], d = dst[e];
#pragma unroll
  for (int h = 0; h < Hh; ++h) {
    float le = 0.f;
#pragma unroll
    for (int f = 0; f < FE; ++f) le += eav[f] * M[f * Hh + h];
    lebuf[e * Hh + h] = le;
    atomicMax(&mEnc[d * Hh + h], encf(lrelu(lsb[s * Hh + h] + ldb[d * Hh + h] + le)));
  }
}

// ---------------- edge pass B: exp + unnormalized aggregate (wave per edge) ----------------
__global__ void k_edge_accum(const int* __restrict__ src, const int* __restrict__ dst,
                             const float* __restrict__ lebuf, const float* __restrict__ lsb,
                             const float* __restrict__ ldb, const unsigned* __restrict__ mEnc,
                             const float* __restrict__ hp, float* __restrict__ den,
                             float* __restrict__ acc, int E) {
  int e = blockIdx.x * 8 + (threadIdx.x >> 5);
  int lane = threadIdx.x & 31;
  if (e >= E) return;
  int s = src[e], d = dst[e];
  float ex[Hh];
#pragma unroll
  for (int h = 0; h < Hh; ++h) {
    float lg = lrelu(lsb[s * Hh + h] + ldb[d * Hh + h] + lebuf[e * Hh + h]);
    ex[h] = __expf(lg - decf(mEnc[d * Hh + h]));
  }
  if (lane < Hh) atomAddF(&den[d * Hh + lane], ex[lane]);
  const float* hs = hp + (size_t)s * HCd;
  float* ao = acc + (size_t)d * HCd;
#pragma unroll
  for (int h = 0; h < Hh; ++h) {
    int c = h * Cc + lane;
    atomAddF(&ao[c], ex[h] * hs[c]);
  }
}

// ---------------- self-loop accumulate (wave per node) ----------------
__global__ void k_self_accum(const float* __restrict__ leself, const float* __restrict__ lsb,
                             const float* __restrict__ ldb, const unsigned* __restrict__ mEnc,
                             const float* __restrict__ hp, float* __restrict__ den,
                             float* __restrict__ acc, int N) {
  int n = blockIdx.x * 8 + (threadIdx.x >> 5);
  int lane = threadIdx.x & 31;
  if (n >= N) return;
  float ex[Hh];
#pragma unroll
  for (int h = 0; h < Hh; ++h) {
    float lg = lrelu(lsb[n * Hh + h] + ldb[n * Hh + h] + leself[h]);
    ex[h] = __expf(lg - decf(mEnc[n * Hh + h]));
  }
  if (lane < Hh) atomAddF(&den[n * Hh + lane], ex[lane]);
  const float* hs = hp + (size_t)n * HCd;
  float* ao = acc + (size_t)n * HCd;
#pragma unroll
  for (int h = 0; h < Hh; ++h) {
    int c = h * Cc + lane;
    atomAddF(&ao[c], ex[h] * hs[c]);
  }
}

// ---------------- finalize: divide by den, +bias, optional ELU ----------------
__global__ void k_finalize(float* __restrict__ acc, const float* __restrict__ den,
                           const float* __restrict__ bias, int N, int doElu) {
  int t = blockIdx.x * blockDim.x + threadIdx.x;
  if (t >= N * HCd) return;
  int n = t >> 7, col = t & 127;
  float v = acc[t] / (den[n * Hh + (col >> 5)] + 1e-16f) + bias[col];
  if (doElu) v = v > 0.f ? v : (__expf(v) - 1.0f);
  acc[t] = v;
}

// ---------------- graph pooling ----------------
__global__ void k_group_accum(const float* __restrict__ h, const int* __restrict__ batch,
                              float* __restrict__ gsum, float* __restrict__ gcnt, int N) {
  int n = blockIdx.x * 8 + (threadIdx.x >> 5);
  int lane = threadIdx.x & 31;
  if (n >= N) return;
  int g = batch[n];
  const float* r = h + (size_t)n * HCd;
#pragma unroll
  for (int j = 0; j < 4; ++j) atomAddF(&gsum[g * HCd + j * 32 + lane], r[j * 32 + lane]);
  if (lane == 0) atomAddF(&gcnt[g], 1.0f);
}

__launch_bounds__(128)
__global__ void k_class_head(const float* __restrict__ gsum, const float* __restrict__ gcnt,
                             const float* __restrict__ Wc1, const float* __restrict__ bc1,
                             const float* __restrict__ Wc2, const float* __restrict__ bc2,
                             float* __restrict__ out) {
  __shared__ float gemb[HCd];
  __shared__ float hid[HCd];
  int g = blockIdx.x, t = threadIdx.x;
  float cnt = fmaxf(gcnt[g], 1.0f);
  gemb[t] = gsum[g * HCd + t] / cnt;
  __syncthreads();
  float a = bc1[t];
  for (int k = 0; k < HCd; ++k) a += gemb[k] * Wc1[k * HCd + t];
  hid[t] = fmaxf(a, 0.f);
  __syncthreads();
  if (t < NCLS) {
    float o = bc2[t];
    for (int k = 0; k < HCd; ++k) o += hid[k] * Wc2[k * NCLS + t];
    out[g * NCLS + t] = o;
  }
}

__global__ void k_loc2(const float* __restrict__ tmp64, const float* __restrict__ Wl2,
                       const float* __restrict__ bl2, float* __restrict__ out, int N) {
  int n = blockIdx.x * blockDim.x + threadIdx.x;
  if (n >= N) return;
  const float* r = tmp64 + (size_t)n * 64;
  float o = bl2[0];
#pragma unroll 8
  for (int k = 0; k < 64; ++k) o += r[k] * Wl2[k];
  out[n] = o;
}

// =======================================================================
extern "C" void kernel_launch(void* const* d_in, const int* in_sizes, int n_in,
                              void* d_out, int out_size, void* d_ws, size_t ws_size,
                              hipStream_t stream) {
  const float* x     = (const float*)d_in[0];
  const int*   ei    = (const int*)d_in[1];
  const float* ea    = (const float*)d_in[2];
  const int*   batch = (const int*)d_in[3];
  const int N = in_sizes[0] / 64;
  const int E = in_sizes[1] / 2;
  const int* src = ei;
  const int* dst = ei + E;

  const float* Wm[3]  = {(const float*)d_in[4],  (const float*)d_in[10], (const float*)d_in[16]};
  const float* aSm[3] = {(const float*)d_in[5],  (const float*)d_in[11], (const float*)d_in[17]};
  const float* aDm[3] = {(const float*)d_in[6],  (const float*)d_in[12], (const float*)d_in[18]};
  const float* Wem[3] = {(const float*)d_in[7],  (const float*)d_in[13], (const float*)d_in[19]};
  const float* aEm[3] = {(const float*)d_in[8],  (const float*)d_in[14], (const float*)d_in[20]};
  const float* bm[3]  = {(const float*)d_in[9],  (const float*)d_in[15], (const float*)d_in[21]};
  const float* Wc1 = (const float*)d_in[22]; const float* bc1 = (const float*)d_in[23];
  const float* Wc2 = (const float*)d_in[24]; const float* bc2 = (const float*)d_in[25];
  const float* Wl1 = (const float*)d_in[26]; const float* bl1 = (const float*)d_in[27];
  const float* Wl2 = (const float*)d_in[28]; const float* bl2 = (const float*)d_in[29];
  float* out = (float*)d_out;

  // ---- workspace carve ----
  char* wp = (char*)d_ws;
  auto carve = [&](size_t bytes) -> void* {
    void* p = (void*)wp;
    wp += (bytes + 255) & ~(size_t)255;
    return p;
  };
  float*    hA     = (float*)carve((size_t)N * HCd * 4);   // layer output / aggregate acc
  float*    hB     = (float*)carve((size_t)N * HCd * 4);   // projected h (gather source)
  float*    lsb    = (float*)carve((size_t)N * Hh * 4);
  float*    ldb    = (float*)carve((size_t)N * Hh * 4);
  float*    den    = (float*)carve((size_t)N * Hh * 4);
  unsigned* mEnc   = (unsigned*)carve((size_t)N * Hh * 4);
  size_t leElems   = (size_t)E * Hh > (size_t)N * 64 ? (size_t)E * Hh : (size_t)N * 64;
  float*    lebuf  = (float*)carve(leElems * 4);           // aliased later as tmp64 for loc head
  float*    gsum   = (float*)carve(NGRP * HCd * 4);
  float*    gcnt   = (float*)carve(NGRP * 4);
  float*    meanS  = (float*)carve(FE * 4);
  float*    Mfold  = (float*)carve(FE * Hh * 4);
  float*    leself = (float*)carve(Hh * 4);
  unsigned short* wbf = (unsigned short*)carve(HCd * HCd * 2);  // transposed bf16 weights
  float*    tmp64  = lebuf;

  const int TB = 256;
  auto blocks = [](long n, int tb) { return (int)((n + tb - 1) / tb); };

  // ---- edge_attr column mean (sums; divided inside k_prep) ----
  k_zero<<<1, 32, 0, stream>>>(meanS, FE);
  k_colsum16<<<256, TB, 0, stream>>>(ea, meanS, E);

  const float* cur = x;   // layer input
  for (int L = 0; L < 3; ++L) {
    const int KIN = (L == 0) ? 64 : 128;
    // fold edge weights, convert node weights to transposed bf16
    k_prep<<<1, 64, 0, stream>>>(Wem[L], aEm[L], meanS, Mfold, leself, E);
    k_wcvt_t<<<blocks((long)KIN * HCd, TB), TB, 0, stream>>>(Wm[L], wbf, KIN, HCd);
    // project:  hB = cur @ W   (WMMA bf16)
    if (L == 0)
      k_gemm<64, 128, false, false><<<(N + 15) / 16, 256, 0, stream>>>(cur, wbf, nullptr, hB, N);
    else
      k_gemm<128, 128, false, false><<<(N + 15) / 16, 256, 0, stream>>>(cur, wbf, nullptr, hB, N);
    // reset aggregation state
    k_zero<<<blocks((long)N * HCd, TB), TB, 0, stream>>>(hA, N * HCd);
    k_zero<<<blocks((long)N * Hh, TB), TB, 0, stream>>>(den, N * Hh);
    k_zero<<<blocks((long)N * Hh, TB), TB, 0, stream>>>((float*)mEnc, N * Hh);
    // alphas + softmax-max
    k_node_alpha<<<blocks((long)N * Hh, TB), TB, 0, stream>>>(hB, aSm[L], aDm[L], leself, lsb, ldb, mEnc, N);
    k_edge_max<<<blocks(E, TB), TB, 0, stream>>>(src, dst, ea, Mfold, lsb, ldb, lebuf, mEnc, E);
    // exp + unnormalized aggregation
    k_edge_accum<<<(E + 7) / 8, TB, 0, stream>>>(src, dst, lebuf, lsb, ldb, mEnc, hB, den, hA, E);
    k_self_accum<<<(N + 7) / 8, TB, 0, stream>>>(leself, lsb, ldb, mEnc, hB, den, hA, N);
    // normalize + bias (+ELU for layers 0,1)
    k_finalize<<<blocks((long)N * HCd, TB), TB, 0, stream>>>(hA, den, bm[L], N, (L < 2) ? 1 : 0);
    cur = hA;
  }

  // ---- node-level head: relu(h@Wl1+bl1) @ Wl2 + bl2 ----
  k_wcvt_t<<<blocks((long)HCd * 64, TB), TB, 0, stream>>>(Wl1, wbf, HCd, 64);
  k_gemm<128, 64, true, true><<<(N + 15) / 16, 128, 0, stream>>>(hA, wbf, bl1, tmp64, N);
  k_loc2<<<blocks(N, TB), TB, 0, stream>>>(tmp64, Wl2, bl2, out + NGRP * NCLS, N);

  // ---- graph-level head ----
  k_zero<<<blocks(NGRP * HCd, TB), TB, 0, stream>>>(gsum, NGRP * HCd);
  k_zero<<<1, NGRP, 0, stream>>>(gcnt, NGRP);
  k_group_accum<<<(N + 7) / 8, TB, 0, stream>>>(hA, batch, gsum, gcnt, N);
  k_class_head<<<NGRP, HCd, 0, stream>>>(gsum, gcnt, Wc1, bc1, Wc2, bc2, out);
}
